// SelfAttention_55920474193996
// MI455X (gfx1250) — matrix operand order repaired and checked by reading, not compiled, not run on hardware
//
#include <hip/hip_runtime.h>
#include <math.h>

typedef __attribute__((ext_vector_type(8)))  float        v8f;
typedef __attribute__((ext_vector_type(2)))  float        v2f;
typedef __attribute__((ext_vector_type(4)))  float        f4;
typedef __attribute__((ext_vector_type(16))) __bf16       v16bf;
typedef __attribute__((ext_vector_type(4)))  unsigned int u4v;

#define B_  2
#define T_  2048
#define D_  2048
#define H_  16
#define HD_ 128
#define N3_ (3 * D_) // 6144

__device__ __forceinline__ v8f wmma_f32(v2f a, v2f b, v8f c) {
    return __builtin_amdgcn_wmma_f32_16x16x4_f32(
        false, a, false, b, (short)0, c, false, false);
}
__device__ __forceinline__ v8f wmma_bf16(v16bf a, v16bf b, v8f c) {
    return __builtin_amdgcn_wmma_f32_16x16x32_bf16(
        false, a, false, b, (short)0, c, false, false);
}
// Load a 16-element bf16 fragment (8 VGPRs) as two 16B LDS loads.
__device__ __forceinline__ v16bf ld_frag(const __bf16* p0, const __bf16* p1) {
    union { u4v u[2]; v16bf v; } f;
    f.u[0] = *(const u4v*)p0;
    f.u[1] = *(const u4v*)p1;
    return f.v;
}
// Async DMA: global -> LDS, 16B per lane, tracked by ASYNCcnt.
__device__ __forceinline__ void async_ld_b128(void* lds, const void* gaddr) {
    unsigned lds_off = (unsigned)(unsigned long long)lds;
    asm volatile("global_load_async_to_lds_b128 %0, %1, off"
                 :: "v"(lds_off), "v"(gaddr) : "memory");
}
#define WAIT_ASYNC_8 asm volatile("s_wait_asynccnt 8" ::: "memory")
#define WAIT_ASYNC_0 asm volatile("s_wait_asynccnt 0" ::: "memory")

__device__ __forceinline__ void split_store(float x, __bf16* hi, __bf16* lo, size_t i) {
    __bf16 h = (__bf16)x;
    hi[i] = h;
    lo[i] = (__bf16)(x - (float)h);
}

// ---------------------------------------------------------------------------
// Pre-pass A: elementwise split f32 -> bf16 hi/lo (for x).
// ---------------------------------------------------------------------------
__global__ __launch_bounds__(256) void split_f32_kernel(
    const float* __restrict__ in, __bf16* __restrict__ hi,
    __bf16* __restrict__ lo, int n)
{
    int i = (blockIdx.x * 256 + threadIdx.x) * 4;
    if (i + 3 < n) {
        f4 v = *(const f4*)(in + i);
#pragma unroll
        for (int j = 0; j < 4; j++) split_store(v[j], hi, lo, i + j);
    }
}

// ---------------------------------------------------------------------------
// Pre-pass B: transpose+split f32 [R][C] -> bf16 hi/lo [C][R]
// (B operands in [N][K] = native 16-bit WMMA B-fragment layout).
// ---------------------------------------------------------------------------
__global__ __launch_bounds__(256) void transpose_split_kernel(
    const float* __restrict__ in, __bf16* __restrict__ ohi,
    __bf16* __restrict__ olo, int R, int C)
{
    __shared__ float tile[32][33];
    int tx = threadIdx.x & 31, ty = threadIdx.x >> 5;
    int r0 = blockIdx.y * 32, c0 = blockIdx.x * 32;
#pragma unroll
    for (int i = ty; i < 32; i += 8)
        tile[i][tx] = in[(size_t)(r0 + i) * C + c0 + tx];
    __syncthreads();
#pragma unroll
    for (int i = ty; i < 32; i += 8)
        split_store(tile[tx][i], ohi, olo, (size_t)(c0 + i) * R + r0 + tx);
}

// ---------------------------------------------------------------------------
// Kernel 1: qkv = x @ qkv_w via 3-term bf16 split-WMMA; fused RoPE+RMSNorm.
// A = x hi/lo [M][K] bf16; B = qkv_w^T hi/lo [N][K] bf16. 128x128 tile,
// K-step 32, double-buffered async staging (8 async ops/wave per tile).
// ---------------------------------------------------------------------------
__global__ __launch_bounds__(256) void qkv_gemm_kernel(
    const __bf16* __restrict__ Ahi_g, const __bf16* __restrict__ Alo_g,
    const __bf16* __restrict__ Bhi_g, const __bf16* __restrict__ Blo_g,
    const int* __restrict__ positions,
    const float* __restrict__ qnw, const float* __restrict__ knw,
    __bf16* __restrict__ qhi, __bf16* __restrict__ qlo,
    __bf16* __restrict__ khi, __bf16* __restrict__ klo,
    float* __restrict__ vout)
{
    extern __shared__ char smemc[];
    // plane layout: 8 planes of [128][40] bf16 (80B row stride)
    __bf16* AH0 = (__bf16*)smemc;
    __bf16* AL0 = AH0 + 128 * 40;
    __bf16* BH0 = AL0 + 128 * 40;
    __bf16* BL0 = BH0 + 128 * 40;
    __bf16* AH1 = BL0 + 128 * 40;
    __bf16* AL1 = AH1 + 128 * 40;
    __bf16* BH1 = AL1 + 128 * 40;
    __bf16* BL1 = BH1 + 128 * 40;
    float*  Cs  = (float*)smemc; // [128][132] f32 alias for epilogue

    const int tid  = threadIdx.x;
    const int lane = tid & 31;
    const int wv   = tid >> 5;
    const int wm   = wv & 3;
    const int wn   = wv >> 2;
    const int l16  = lane & 15;
    const int hiA  = (lane >> 4) << 3;  // A frag: +8 elems for lanes 16..31
    const int hiB  = (lane >> 4) << 3;  // B frag: +8 elems for lanes 16..31
    const int hi8  = (lane >> 4) << 3;  // C row half

    const int nb  = blockIdx.x;  // 0..47
    const int mb  = blockIdx.y;  // 0..31
    const int sel = nb >> 4;     // 0=q 1=k 2=v
    const int h   = nb & 15;

    v8f acc[2][4];
    const v8f vz = {0.f,0.f,0.f,0.f,0.f,0.f,0.f,0.f};
#pragma unroll
    for (int i = 0; i < 2; i++)
#pragma unroll
        for (int j = 0; j < 4; j++) acc[i][j] = vz;

    auto issue_tile = [&](int kk, __bf16* dAH, __bf16* dAL,
                          __bf16* dBH, __bf16* dBL) {
#pragma unroll
        for (int j = 0; j < 2; j++) {
            int li = tid + j * 256;
            int r = li >> 2, c8 = (li & 3) << 3;
            async_ld_b128(dAH + r * 40 + c8,
                          Ahi_g + (size_t)(mb * 128 + r) * D_ + kk + c8);
        }
#pragma unroll
        for (int j = 0; j < 2; j++) {
            int li = tid + j * 256;
            int r = li >> 2, c8 = (li & 3) << 3;
            async_ld_b128(dAL + r * 40 + c8,
                          Alo_g + (size_t)(mb * 128 + r) * D_ + kk + c8);
        }
#pragma unroll
        for (int j = 0; j < 2; j++) {
            int li = tid + j * 256;
            int r = li >> 2, c8 = (li & 3) << 3;
            async_ld_b128(dBH + r * 40 + c8,
                          Bhi_g + (size_t)(nb * 128 + r) * D_ + kk + c8);
        }
#pragma unroll
        for (int j = 0; j < 2; j++) {
            int li = tid + j * 256;
            int r = li >> 2, c8 = (li & 3) << 3;
            async_ld_b128(dBL + r * 40 + c8,
                          Blo_g + (size_t)(nb * 128 + r) * D_ + kk + c8);
        }
    };

    issue_tile(0, AH0, AL0, BH0, BL0);

    for (int kk = 0; kk < D_; kk += 32) {
        const int p = (kk >> 5) & 1;
        const __bf16* cAH = p ? AH1 : AH0;
        const __bf16* cAL = p ? AL1 : AL0;
        const __bf16* cBH = p ? BH1 : BH0;
        const __bf16* cBL = p ? BL1 : BL0;
        if (kk + 32 < D_) {
            issue_tile(kk + 32, p ? AH0 : AH1, p ? AL0 : AL1,
                       p ? BH0 : BH1, p ? BL0 : BL1);
            WAIT_ASYNC_8;
        } else {
            WAIT_ASYNC_0;
        }
        __syncthreads();

        v16bf ah[2], al[2];
#pragma unroll
        for (int i = 0; i < 2; i++) {
            int row = wm * 32 + i * 16 + l16;
            const __bf16* pa = cAH + row * 40 + hiA;
            const __bf16* pl = cAL + row * 40 + hiA;
            ah[i] = ld_frag(pa, pa + 16);
            al[i] = ld_frag(pl, pl + 16);
        }
#pragma unroll
        for (int j = 0; j < 4; j++) {
            int n = wn * 64 + j * 16 + l16;
            const __bf16* pb = cBH + n * 40 + hiB;
            const __bf16* pq = cBL + n * 40 + hiB;
            v16bf bh = ld_frag(pb, pb + 16);
            v16bf bl = ld_frag(pq, pq + 16);
#pragma unroll
            for (int i = 0; i < 2; i++) {
                acc[i][j] = wmma_bf16(ah[i], bh, acc[i][j]);
                acc[i][j] = wmma_bf16(ah[i], bl, acc[i][j]);
                acc[i][j] = wmma_bf16(al[i], bh, acc[i][j]);
            }
        }
        __syncthreads();
    }

#pragma unroll
    for (int i = 0; i < 2; i++)
#pragma unroll
        for (int j = 0; j < 4; j++) {
            int row = wm * 32 + i * 16 + hi8;
            int col = wn * 64 + j * 16 + l16;
#pragma unroll
            for (int v = 0; v < 8; v++)
                Cs[(row + v) * 132 + col] = acc[i][j][v];
        }
    __syncthreads();

    if (tid < 128) {
        int row  = tid;
        int grow = mb * 128 + row;
        int b    = grow / T_;
        int t    = grow % T_;
        size_t base = (((size_t)(b * H_ + h)) * T_ + t) * HD_;
        float* crow = Cs + row * 132;

        if (sel == 2) {
            for (int i = 0; i < HD_; i++) vout[base + i] = crow[i];
        } else {
            const float* nw = (sel == 0) ? qnw : knw;
            __bf16* dh = (sel == 0) ? qhi : khi;
            __bf16* dl = (sel == 0) ? qlo : klo;
            float pos = (float)positions[b * T_ + t];
            float ss = 0.f;
            for (int i = 0; i < 64; i++) { // RoPE
                float invf = __expf(-(float)i * 0.14391156831f);
                float sa, ca;
                __sincosf(pos * invf, &sa, &ca);
                float x1 = crow[i], x2 = crow[i + 64];
                float y1 = x1 * ca - x2 * sa;
                float y2 = x2 * ca + x1 * sa;
                crow[i] = y1; crow[i + 64] = y2;
                ss += y1 * y1 + y2 * y2;
            }
            float sc = rsqrtf(ss * (1.0f / 128.0f) + 1e-6f); // RMSNorm
            for (int i = 0; i < HD_; i++)
                split_store(crow[i] * sc * nw[i], dh, dl, base + i);
        }
    }
}

// ---------------------------------------------------------------------------
// Kernel 2: flash attention. S = Q K^T via 3-term bf16 split-WMMA
// (K stored [key][hd] == B-fragment layout for K^T). Softmax f32.
// O += P V via f32 WMMA. Async DMA: V stream hidden behind S + softmax.
// ---------------------------------------------------------------------------
__global__ __launch_bounds__(256) void attention_kernel(
    const __bf16* __restrict__ qhi_g, const __bf16* __restrict__ qlo_g,
    const __bf16* __restrict__ khi_g, const __bf16* __restrict__ klo_g,
    const float* __restrict__ vg,
    __bf16* __restrict__ ohi_g, __bf16* __restrict__ olo_g)
{
    extern __shared__ char smemc[];
    __bf16* QH = (__bf16*)smemc;        // [128][136]
    __bf16* QL = QH + 128 * 136;        // [128][136]
    __bf16* KH = QL + 128 * 136;        // [64][136]
    __bf16* KL = KH + 64 * 136;         // [64][136]
    float*  Vs = (float*)(KL + 64 * 136);   // [64][132]
    float*  Ss = Vs + 64 * 132;             // [128][68]

    const int tid   = threadIdx.x;
    const int lane  = tid & 31;
    const int wv    = tid >> 5;
    const int l16   = lane & 15;
    const int khalf = (lane >> 4) << 1;
    const int hiA   = (lane >> 4) << 3;
    const int hiB   = (lane >> 4) << 3;
    const int hi8   = (lane >> 4) << 3;

    const int qb = blockIdx.x; // 0..15
    const int bh = blockIdx.y; // 0..31

    const size_t qoff = (((size_t)bh) * T_ + qb * 128) * HD_;
    const __bf16* Qh = qhi_g + qoff;
    const __bf16* Ql = qlo_g + qoff;
    const __bf16* Kh = khi_g + ((size_t)bh) * T_ * HD_;
    const __bf16* Kl = klo_g + ((size_t)bh) * T_ * HD_;
    const float*  Vg = vg + ((size_t)bh) * T_ * HD_;

#pragma unroll
    for (int j = 0; j < 8; j++) { // Q hi 128x128 bf16
        int li = tid + j * 256;
        int r = li >> 4, c8 = (li & 15) << 3;
        async_ld_b128(QH + r * 136 + c8, Qh + (size_t)r * HD_ + c8);
    }
#pragma unroll
    for (int j = 0; j < 8; j++) { // Q lo
        int li = tid + j * 256;
        int r = li >> 4, c8 = (li & 15) << 3;
        async_ld_b128(QL + r * 136 + c8, Ql + (size_t)r * HD_ + c8);
    }

    const v8f vz = {0.f,0.f,0.f,0.f,0.f,0.f,0.f,0.f};
    v8f oacc[8];
#pragma unroll
    for (int t = 0; t < 8; t++) oacc[t] = vz;
    float mrow[8], lrow[8];
#pragma unroll
    for (int v = 0; v < 8; v++) { mrow[v] = -3.0e38f; lrow[v] = 0.f; }

    const float scale = 0.08838834764831845f; // 1/sqrt(128)
    const int nkb = 2 * qb + 2;

    for (int j = 0; j < nkb; j++) {
#pragma unroll
        for (int jj = 0; jj < 4; jj++) { // K hi 64x128 bf16
            int li = tid + jj * 256;
            int r = li >> 4, c8 = (li & 15) << 3;
            async_ld_b128(KH + r * 136 + c8,
                          Kh + (size_t)(j * 64 + r) * HD_ + c8);
        }
#pragma unroll
        for (int jj = 0; jj < 4; jj++) { // K lo
            int li = tid + jj * 256;
            int r = li >> 4, c8 = (li & 15) << 3;
            async_ld_b128(KL + r * 136 + c8,
                          Kl + (size_t)(j * 64 + r) * HD_ + c8);
        }
#pragma unroll
        for (int jj = 0; jj < 8; jj++) { // V 64x128 f32
            int li = tid + jj * 256;
            int r = li >> 5, c4 = (li & 31) << 2;
            async_ld_b128(Vs + r * 132 + c4,
                          Vg + (size_t)(j * 64 + r) * HD_ + c4);
        }
        WAIT_ASYNC_8; // Q (1st iter) + K planes resident; V still streaming
        __syncthreads();

        v8f sacc[4];
#pragma unroll
        for (int ct = 0; ct < 4; ct++) sacc[ct] = vz;
#pragma unroll
        for (int k0 = 0; k0 < 128; k0 += 32) { // S = Q K^T, 3-term bf16
            int row = wv * 16 + l16;
            const __bf16* pa = QH + row * 136 + k0 + hiA;
            const __bf16* pl = QL + row * 136 + k0 + hiA;
            v16bf ah = ld_frag(pa, pa + 16);
            v16bf al = ld_frag(pl, pl + 16);
#pragma unroll
            for (int ct = 0; ct < 4; ct++) {
                int key = ct * 16 + l16;
                const __bf16* pb = KH + key * 136 + k0 + hiB;
                const __bf16* pq = KL + key * 136 + k0 + hiB;
                v16bf bhf = ld_frag(pb, pb + 16);
                v16bf blf = ld_frag(pq, pq + 16);
                sacc[ct] = wmma_bf16(ah, bhf, sacc[ct]);
                sacc[ct] = wmma_bf16(ah, blf, sacc[ct]);
                sacc[ct] = wmma_bf16(al, bhf, sacc[ct]);
            }
        }

        const int grow_base = qb * 128 + wv * 16 + hi8;
        float pmax[8];
#pragma unroll
        for (int v = 0; v < 8; v++) pmax[v] = -3.0e38f;
#pragma unroll
        for (int ct = 0; ct < 4; ct++) {
            int gcol = j * 64 + ct * 16 + l16;
#pragma unroll
            for (int v = 0; v < 8; v++) {
                float s = sacc[ct][v] * scale;
                s = (gcol <= grow_base + v) ? s : -3.0e38f;
                sacc[ct][v] = s;
                pmax[v] = fmaxf(pmax[v], s);
            }
        }
#pragma unroll
        for (int v = 0; v < 8; v++) {
            float m = pmax[v];
            m = fmaxf(m, __shfl_xor(m, 1, 16));
            m = fmaxf(m, __shfl_xor(m, 2, 16));
            m = fmaxf(m, __shfl_xor(m, 4, 16));
            m = fmaxf(m, __shfl_xor(m, 8, 16));
            pmax[v] = m;
        }
        float sf[8], psum[8];
#pragma unroll
        for (int v = 0; v < 8; v++) {
            float mnew = fmaxf(mrow[v], pmax[v]);
            sf[v]   = __expf(mrow[v] - mnew);
            mrow[v] = mnew;
            psum[v] = 0.f;
        }
#pragma unroll
        for (int ct = 0; ct < 4; ct++) {
#pragma unroll
            for (int v = 0; v < 8; v++) {
                float pe = __expf(sacc[ct][v] - mrow[v]);
                psum[v] += pe;
                Ss[(wv * 16 + hi8 + v) * 68 + ct * 16 + l16] = pe;
            }
        }
#pragma unroll
        for (int v = 0; v < 8; v++) {
            float s = psum[v];
            s += __shfl_xor(s, 1, 16);
            s += __shfl_xor(s, 2, 16);
            s += __shfl_xor(s, 4, 16);
            s += __shfl_xor(s, 8, 16);
            lrow[v] = lrow[v] * sf[v] + s;
        }
#pragma unroll
        for (int t = 0; t < 8; t++)
#pragma unroll
            for (int v = 0; v < 8; v++) oacc[t][v] *= sf[v];

        WAIT_ASYNC_0; // V resident
        __syncthreads();

#pragma unroll 4
        for (int k4 = 0; k4 < 64; k4 += 4) { // O += P V (f32 WMMA)
            v2f a;
            int row = wv * 16 + l16;
            a.x = Ss[row * 68 + k4 + khalf];
            a.y = Ss[row * 68 + k4 + khalf + 1];
#pragma unroll
            for (int ct = 0; ct < 8; ct++) {
                v2f b;
                int col = ct * 16 + l16;
                b.x = Vs[(k4 + khalf) * 132 + col];
                b.y = Vs[(k4 + khalf + 1) * 132 + col];
                oacc[ct] = wmma_f32(a, b, oacc[ct]);
            }
        }
        __syncthreads(); // done with KH/KL/Vs/Ss before next DMA overwrites
    }

#pragma unroll
    for (int v = 0; v < 8; v++) lrow[v] = 1.0f / lrow[v];
    __bf16* Oh = ohi_g + qoff;
    __bf16* Ol = olo_g + qoff;
#pragma unroll
    for (int ct = 0; ct < 8; ct++) {
        int col = ct * 16 + l16;
#pragma unroll
        for (int v = 0; v < 8; v++) {
            int row = wv * 16 + hi8 + v;
            split_store(oacc[ct][v] * lrow[v], Oh, Ol, (size_t)row * HD_ + col);
        }
    }
}

// ---------------------------------------------------------------------------
// Kernel 3: out = attn_out @ out_w via 3-term bf16 split-WMMA.
// A = o hi/lo ([B,H,T,HD] gathered); B = out_w^T hi/lo [N][K].
// ---------------------------------------------------------------------------
__global__ __launch_bounds__(256) void out_proj_kernel(
    const __bf16* __restrict__ Ahi_g, const __bf16* __restrict__ Alo_g,
    const __bf16* __restrict__ Bhi_g, const __bf16* __restrict__ Blo_g,
    float* __restrict__ out)
{
    extern __shared__ char smemc[];
    __bf16* AH0 = (__bf16*)smemc;
    __bf16* AL0 = AH0 + 128 * 40;
    __bf16* BH0 = AL0 + 128 * 40;
    __bf16* BL0 = BH0 + 128 * 40;
    __bf16* AH1 = BL0 + 128 * 40;
    __bf16* AL1 = AH1 + 128 * 40;
    __bf16* BH1 = AL1 + 128 * 40;
    __bf16* BL1 = BH1 + 128 * 40;

    const int tid  = threadIdx.x;
    const int lane = tid & 31;
    const int wv   = tid >> 5;
    const int wm   = wv & 3;
    const int wn   = wv >> 2;
    const int l16  = lane & 15;
    const int hiA  = (lane >> 4) << 3;
    const int hiB  = (lane >> 4) << 3;
    const int hi8  = (lane >> 4) << 3;

    const int nb = blockIdx.x; // 0..15
    const int mb = blockIdx.y; // 0..31

    v8f acc[2][4];
    const v8f vz = {0.f,0.f,0.f,0.f,0.f,0.f,0.f,0.f};
#pragma unroll
    for (int i = 0; i < 2; i++)
#pragma unroll
        for (int j = 0; j < 4; j++) acc[i][j] = vz;

    auto issue_tile = [&](int kk, __bf16* dAH, __bf16* dAL,
                          __bf16* dBH, __bf16* dBL) {
#pragma unroll
        for (int j = 0; j < 2; j++) { // A hi: gather [B,H,T,HD] -> [B*T][D]
            int li = tid + j * 256;
            int r = li >> 2, c8 = (li & 3) << 3;
            int m = mb * 128 + r;
            int b = m / T_, t = m % T_;
            int kcol = kk + c8;
            int hh = kcol >> 7, hd = kcol & 127;
            size_t src = (((size_t)(b * H_ + hh)) * T_ + t) * HD_ + hd;
            async_ld_b128(dAH + r * 40 + c8, Ahi_g + src);
        }
#pragma unroll
        for (int j = 0; j < 2; j++) { // A lo
            int li = tid + j * 256;
            int r = li >> 2, c8 = (li & 3) << 3;
            int m = mb * 128 + r;
            int b = m / T_, t = m % T_;
            int kcol = kk + c8;
            int hh = kcol >> 7, hd = kcol & 127;
            size_t src = (((size_t)(b * H_ + hh)) * T_ + t) * HD_ + hd;
            async_ld_b128(dAL + r * 40 + c8, Alo_g + src);
        }
#pragma unroll
        for (int j = 0; j < 2; j++) { // B hi
            int li = tid + j * 256;
            int r = li >> 2, c8 = (li & 3) << 3;
            async_ld_b128(dBH + r * 40 + c8,
                          Bhi_g + (size_t)(nb * 128 + r) * D_ + kk + c8);
        }
#pragma unroll
        for (int j = 0; j < 2; j++) { // B lo
            int li = tid + j * 256;
            int r = li >> 2, c8 = (li & 3) << 3;
            async_ld_b128(dBL + r * 40 + c8,
                          Blo_g + (size_t)(nb * 128 + r) * D_ + kk + c8);
        }
    };

    issue_tile(0, AH0, AL0, BH0, BL0);

    for (int kk = 0; kk < D_; kk += 32) {
        const int p = (kk >> 5) & 1;
        const __bf16* cAH = p ? AH1 : AH0;
        const __bf16* cAL = p ? AL1 : AL0;
        const __bf16* cBH = p ? BH1 : BH0;
        const __bf16* cBL = p ? BL1 : BL0;
        if (kk + 32 < D_) {
            issue_tile(kk + 32, p ? AH0 : AH1, p ? AL0 : AL1,
                       p ? BH0 : BH1, p ? BL0 : BL1);
            WAIT_ASYNC_8;
        } else {
            WAIT_ASYNC_0;
        }
        __syncthreads();

        v16bf ah[2], al[2];
#pragma unroll
        for (int i = 0; i < 2; i++) {
            int row = wm * 32 + i * 16 + l16;
            const __bf16* pa = cAH + row * 40 + hiA;
            const __bf16* pl = cAL + row * 40 + hiA;
            ah[i] = ld_frag(pa, pa + 16);
            al[i] = ld_frag(pl, pl + 16);
        }
#pragma unroll
        for (int j = 0; j < 4; j++) {
            int n = wn * 64 + j * 16 + l16;
            const __bf16* pb = cBH + n * 40 + hiB;
            const __bf16* pq = cBL + n * 40 + hiB;
            v16bf bh = ld_frag(pb, pb + 16);
            v16bf bl = ld_frag(pq, pq + 16);
#pragma unroll
            for (int i = 0; i < 2; i++) {
                acc[i][j] = wmma_bf16(ah[i], bh, acc[i][j]);
                acc[i][j] = wmma_bf16(ah[i], bl, acc[i][j]);
                acc[i][j] = wmma_bf16(al[i], bh, acc[i][j]);
            }
        }
        __syncthreads();
    }

#pragma unroll
    for (int i = 0; i < 2; i++)
#pragma unroll
        for (int j = 0; j < 4; j++) {
            int row = mb * 128 + wm * 32 + i * 16 + hi8;
            int col = nb * 128 + wn * 64 + j * 16 + l16;
#pragma unroll
            for (int v = 0; v < 8; v++)
                out[(size_t)(row + v) * D_ + col] = acc[i][j][v];
        }
}

extern "C" void kernel_launch(void* const* d_in, const int* in_sizes, int n_in,
                              void* d_out, int out_size, void* d_ws, size_t ws_size,
                              hipStream_t stream) {
    (void)in_sizes; (void)n_in; (void)out_size; (void)ws_size;
    const float* x         = (const float*)d_in[0];
    // d_in[1] = attn_mask (causal tril) — enforced analytically in-kernel
    const int*   positions = (const int*)d_in[2];
    const float* qkv_w     = (const float*)d_in[3];
    const float* out_w     = (const float*)d_in[4];
    const float* qnw       = (const float*)d_in[5];
    const float* knw       = (const float*)d_in[6];
    float* out = (float*)d_out;

    const size_t SXD = (size_t)B_ * T_ * D_;        // 8388608
    char* p = (char*)d_ws;
    __bf16* qkvwT_hi = (__bf16*)p; p += (size_t)N3_ * D_ * 2;
    __bf16* qkvwT_lo = (__bf16*)p; p += (size_t)N3_ * D_ * 2;
    __bf16* outwT_hi = (__bf16*)p; p += (size_t)D_ * D_ * 2;
    __bf16* outwT_lo = (__bf16*)p; p += (size_t)D_ * D_ * 2;
    __bf16* xhi      = (__bf16*)p; p += SXD * 2;
    __bf16* xlo      = (__bf16*)p; p += SXD * 2;
    __bf16* qhi      = (__bf16*)p; p += SXD * 2;
    __bf16* qlo      = (__bf16*)p; p += SXD * 2;
    __bf16* khi      = (__bf16*)p; p += SXD * 2;
    __bf16* klo      = (__bf16*)p; p += SXD * 2;
    float*  vbuf     = (float*)p;  p += SXD * 4;
    __bf16* ohi      = xhi; // x planes dead after kernel 1
    __bf16* olo      = xlo;

    // Pre-pass: split x; transpose+split weights into [N][K] bf16 planes.
    split_f32_kernel<<<(int)(SXD / 1024), 256, 0, stream>>>(x, xhi, xlo, (int)SXD);
    transpose_split_kernel<<<dim3(N3_ / 32, D_ / 32), 256, 0, stream>>>(
        qkv_w, qkvwT_hi, qkvwT_lo, D_, N3_);
    transpose_split_kernel<<<dim3(D_ / 32, D_ / 32), 256, 0, stream>>>(
        out_w, outwT_hi, outwT_lo, D_, D_);

    const int smemG = 8 * 128 * 40 * 2;                               // 81920 B
    const int smemA = (2 * 128 * 136 + 2 * 64 * 136) * 2
                    + (64 * 132 + 128 * 68) * 4;                      // 173056 B

    qkv_gemm_kernel<<<dim3(48, 32), 256, smemG, stream>>>(
        xhi, xlo, qkvwT_hi, qkvwT_lo, positions, qnw, knw,
        qhi, qlo, khi, klo, vbuf);
    attention_kernel<<<dim3(16, 32), 256, smemA, stream>>>(
        qhi, qlo, khi, klo, vbuf, ohi, olo);
    out_proj_kernel<<<dim3(16, 32), 256, smemG, stream>>>(
        ohi, olo, outwT_hi, outwT_lo, out);
}
